// AttentionNetwork_80384607912453
// MI455X (gfx1250) — matrix operand order repaired and checked by reading, not compiled, not run on hardware
//
#include <hip/hip_runtime.h>

typedef float v8f  __attribute__((ext_vector_type(8)));
typedef __bf16 v16bf __attribute__((ext_vector_type(16)));
typedef unsigned int v4u __attribute__((ext_vector_type(4)));
typedef int v8i __attribute__((ext_vector_type(8)));
typedef int v4i __attribute__((ext_vector_type(4)));

#define Bk 4
#define Fk 256
#define Tk 8192
#define Hk 1024
#define Ck 128
#define TB 32
#define NSPLIT 8            // t-splits in stage 2

#define SEQ_S 40            // fp32 seq tile row stride (32 data + 8 pad dwords; TDM pads)
#define SEQH_S 272          // bf16 A-layout seq row stride (256 + 16)
#define ZH_S  1040          // bf16 A-layout z row stride (1024 + 16)
#define LOG_S 132           // fp32 logits row stride
#define TFRAG_S 48          // bf16 K=32 fragment row stride (32 + 16)

#define USE_TDM 1

static __device__ __forceinline__ __bf16 tobf(float x) { return (__bf16)x; }

static __device__ __forceinline__ v8f wmma_bf16(v16bf a, v16bf b, v8f c) {
    return __builtin_amdgcn_wmma_f32_16x16x32_bf16(false, a, false, b, (short)0, c,
                                                   false, false);
}

static __device__ __forceinline__ v8f splat8(float x) {
    v8f c;
#pragma unroll
    for (int r = 0; r < 8; ++r) c[r] = x;
    return c;
}

// bf16 16x16x32 A-fragment slot for K index k in [0,32): half*16 + elem
static __device__ __forceinline__ int a_perm(int k) {
    int half = k >> 4;
    int e = k & 15;
    return half * 16 + e;
}

#if USE_TDM
// TDM: load 2D tile [Fk rows x TB floats], row stride Tk floats, into LDS with
// 8-dword padding after every 32 dwords (-> LDS row stride SEQ_S=40 floats).
static __device__ __forceinline__ void tdm_load_seq_tile(const float* gsrc,
                                                         unsigned lds_off) {
    unsigned long long ga = (unsigned long long)(uintptr_t)gsrc;
    v4u g0;
    g0.x = 1u;                                             // count=1, user D#
    g0.y = lds_off;                                        // lds_addr (bytes)
    g0.z = (unsigned)ga;                                   // global_addr lo32
    g0.w = (unsigned)((ga >> 32) & 0x1FFFFFFu) | (2u << 30); // addr hi, type=2
    v8i g1;
    g1[0] = (int)((2u << 16)      // data_size = 4B
                | (1u << 20)      // pad_enable
                | (4u << 22)      // pad_interval = 32 dwords
                | (7u << 25));    // pad_amount  = 8 dwords
    g1[1] = (int)(((unsigned)Tk & 0xFFFFu) << 16);         // tensor_dim0 lo16
    g1[2] = (int)(((unsigned)Tk >> 16) | (((unsigned)Fk & 0xFFFFu) << 16));
    g1[3] = (int)(((unsigned)Fk >> 16) | ((unsigned)TB << 16)); // tile_dim0=32
    g1[4] = (int)Fk;                                       // tile_dim1 = 256
    g1[5] = (int)Tk;                                       // dim0_stride lo32
    g1[6] = 0;
    g1[7] = 0;
    v4i z4 = {0, 0, 0, 0};
    v8i z8 = {0, 0, 0, 0, 0, 0, 0, 0};
    __builtin_amdgcn_tensor_load_to_lds(g0, g1, z4, z4, z8, 0);
}
#endif

// Stage seq tile into s_seqf and convert into bf16 A-fragment layout s_seqh.
static __device__ __forceinline__ void load_convert_seq_tile(
    const float* gsrc, float* s_seqf, __bf16* s_seqh, int tid, int wave) {
#if USE_TDM
    if (wave == 0) {
        tdm_load_seq_tile(gsrc, (unsigned)(uintptr_t)(void*)s_seqf);
        __builtin_amdgcn_s_wait_tensorcnt(0);
    }
#else
    for (int j = tid; j < Fk * (TB / 4); j += 256) {
        int f = j >> 3, tq = j & 7;
        float4 v = *(const float4*)(gsrc + (size_t)f * Tk + tq * 4);
        *(float4*)(&s_seqf[f * SEQ_S + tq * 4]) = v;
    }
#endif
    __syncthreads();
    for (int idx = tid; idx < Fk * TB; idx += 256) {
        int t = idx >> 8, f = idx & 255;
        float v = s_seqf[f * SEQ_S + t];
        s_seqh[t * SEQH_S + (f >> 5) * 32 + a_perm(f & 31)] = tobf(v);
    }
    __syncthreads();
}

// Gather a bf16 B-fragment from an fp32 row-major matrix [K x N] (stride ldn):
// lane n = L&15 takes column col+n, K = kbase + 16*(L>=16) + e.
static __device__ __forceinline__ v16bf load_b_frag_f32(const float* p, int ldn) {
    v16bf b;
#pragma unroll
    for (int e = 0; e < 16; ++e) b[e] = tobf(p[(size_t)e * ldn]);
    return b;
}

// =============== Stage 1: z -> logits -> softmax -> attention ===============
__global__ __launch_bounds__(256, 2) void stage1_attention(
    const float* __restrict__ seq, const float* __restrict__ Wb,
    const float* __restrict__ bb, const float* __restrict__ Wa,
    const float* __restrict__ ba, float* __restrict__ att_out) {
    extern __shared__ char smem[];
    float*  s_seqf  = (float*)smem;                        // 40960 B
    float*  s_logit = (float*)smem;                        // aliases s_seqf
    __bf16* s_seqh  = (__bf16*)(smem + 40960);             // 17408 B
    __bf16* s_zh    = (__bf16*)(smem + 40960 + 17408);     // 66560 B

    const int tid  = threadIdx.x;
    const int wave = tid >> 5;
    const int lane = tid & 31;
    const int m    = lane & 15;
    const int hsel = lane >> 4;
    const int b    = blockIdx.x / (Tk / TB);
    const int t0   = (blockIdx.x % (Tk / TB)) * TB;

    load_convert_seq_tile(seq + (size_t)b * Fk * Tk + t0, s_seqf, s_seqh, tid, wave);

    // ---- Phase 1: z[32,1024] = seq^T @ Wb + bb (bf16 WMMA, K=256) ----
    // Two chunks of 64 h-columns; tile loops fully unrolled so acc stays in VGPRs.
#pragma unroll 1
    for (int ch = 0; ch < 2; ++ch) {
        const int h0c = wave * 128 + ch * 64;
        v8f acc[2][4];
#pragma unroll
        for (int hi = 0; hi < 4; ++hi) {
            v8f c = splat8(bb[h0c + hi * 16 + m]);
            acc[0][hi] = c;
            acc[1][hi] = c;
        }
#pragma unroll 1
        for (int g = 0; g < 8; ++g) {
            v16bf a0 = *(const v16bf*)(s_seqh + m * SEQH_S + g * 32 + hsel * 16);
            v16bf a1 = *(const v16bf*)(s_seqh + (16 + m) * SEQH_S + g * 32 + hsel * 16);
            if (g < 7)
                __builtin_prefetch(&Wb[(size_t)((g + 1) * 32 + lane) * Hk + h0c], 0, 1);
#pragma unroll
            for (int hi = 0; hi < 4; ++hi) {
                v16bf bf = load_b_frag_f32(
                    Wb + (size_t)(g * 32 + hsel * 16) * Hk + h0c + hi * 16 + m, Hk);
                acc[0][hi] = wmma_bf16(a0, bf, acc[0][hi]);
                acc[1][hi] = wmma_bf16(a1, bf, acc[1][hi]);
            }
        }
        // store z as bf16 in logits-A layout: [t][h-group fragment]
#pragma unroll
        for (int ts = 0; ts < 2; ++ts)
#pragma unroll
            for (int hi = 0; hi < 4; ++hi)
#pragma unroll
                for (int r = 0; r < 8; ++r) {
                    int t = ts * 16 + r + hsel * 8;
                    int h = h0c + hi * 16 + m;
                    s_zh[t * ZH_S + (h >> 5) * 32 + a_perm(h & 31)] =
                        tobf(acc[ts][hi][r]);
                }
    }
    __syncthreads();

    // ---- Phase 2: logits[32,128] = z @ Wa + ba (bf16 WMMA, K=1024) ----
    {
        const int ts  = wave & 1;
        const int cq0 = (wave >> 1) * 16;     // c-tiles cq0 and cq0+64
        v8f acc[2];
#pragma unroll
        for (int q = 0; q < 2; ++q) acc[q] = splat8(ba[cq0 + q * 64 + m]);
#pragma unroll 1
        for (int g2 = 0; g2 < 32; ++g2) {
            v16bf a = *(const v16bf*)(s_zh + (ts * 16 + m) * ZH_S + g2 * 32 + hsel * 16);
            if (g2 < 31)
                __builtin_prefetch(&Wa[(size_t)((g2 + 1) * 32 + lane) * Ck], 0, 1);
#pragma unroll
            for (int q = 0; q < 2; ++q) {
                v16bf bf = load_b_frag_f32(
                    Wa + (size_t)(g2 * 32 + hsel * 16) * Ck + cq0 + q * 64 + m, Ck);
                acc[q] = wmma_bf16(a, bf, acc[q]);
            }
        }
        __syncthreads();                      // s_seqf dead -> reuse as s_logit
#pragma unroll
        for (int q = 0; q < 2; ++q)
#pragma unroll
            for (int r = 0; r < 8; ++r)
                s_logit[(ts * 16 + r + hsel * 8) * LOG_S + cq0 + q * 64 + m] = acc[q][r];
    }
    __syncthreads();

    // ---- Softmax over C (8 lanes per row), attention -> d_out ----
    {
        int row = tid >> 3;
        int j   = tid & 7;
        float* rp = s_logit + row * LOG_S + j * 16;
        float v[16];
        float mx = -3.0e38f;
#pragma unroll
        for (int i = 0; i < 4; ++i) {
            float4 q = ((const float4*)rp)[i];
            v[4 * i + 0] = q.x; v[4 * i + 1] = q.y;
            v[4 * i + 2] = q.z; v[4 * i + 3] = q.w;
        }
#pragma unroll
        for (int i = 0; i < 16; ++i) mx = fmaxf(mx, v[i]);
        mx = fmaxf(mx, __shfl_xor(mx, 1, 32));
        mx = fmaxf(mx, __shfl_xor(mx, 2, 32));
        mx = fmaxf(mx, __shfl_xor(mx, 4, 32));
        float s = 0.0f;
#pragma unroll
        for (int i = 0; i < 16; ++i) {
            v[i] = __expf(v[i] - mx);
            s += v[i];
        }
        s += __shfl_xor(s, 1, 32);
        s += __shfl_xor(s, 2, 32);
        s += __shfl_xor(s, 4, 32);
        float rs = 1.0f / (s * (float)Tk);
        float* gp = att_out + ((size_t)b * Tk + t0 + row) * Ck + j * 16;
#pragma unroll
        for (int i = 0; i < 4; ++i) {
            float4 q;
            q.x = v[4 * i + 0] * rs; q.y = v[4 * i + 1] * rs;
            q.z = v[4 * i + 2] * rs; q.w = v[4 * i + 3] * rs;
            ((float4*)gp)[i] = q;
        }
    }
}

// ====== Stage 2: rep[b,h,c] = sum_t z[t,h]*att[t,c], z recomputed bf16 ======
__global__ __launch_bounds__(256, 2) void stage2_rep(
    const float* __restrict__ seq, const float* __restrict__ Wb,
    const float* __restrict__ bb, const float* __restrict__ att,
    float* __restrict__ rep_out) {
    extern __shared__ char smem[];
    float*  s_seqf = (float*)smem;                             // 40960 B
    __bf16* s_seqh = (__bf16*)(smem + 40960);                  // 17408 B
    __bf16* s_zh   = (__bf16*)(smem + 40960 + 17408);          // 12288 B
    __bf16* s_attT = (__bf16*)(smem + 40960 + 17408 + 12288);  // 12288 B

    const int tid  = threadIdx.x;
    const int wave = tid >> 5;
    const int lane = tid & 31;
    const int m    = lane & 15;
    const int hsel = lane >> 4;
    const int wg   = blockIdx.x;
    const int b    = wg >> 6;
    const int hb   = (wg >> 3) & 7;
    const int tsp  = wg & 7;
    const int h0b  = hb * 128;

    v8f acc[8];                 // 128x128 rep tile: wave owns h-tile row `wave`
#pragma unroll
    for (int ci = 0; ci < 8; ++ci) acc[ci] = splat8(0.0f);

    const float* seqb = seq + (size_t)b * Fk * Tk;

#pragma unroll 1
    for (int it = 0; it < (Tk / TB) / NSPLIT; ++it) {          // 32 iterations
        const int t0 = (tsp + it * NSPLIT) * TB;
        __syncthreads();       // protect LDS reuse across iterations
        load_convert_seq_tile(seqb + t0, s_seqf, s_seqh, tid, wave);

        // att tile [32,128] fp32 -> transposed bf16 [c][t]
        for (int idx = tid; idx < TB * Ck; idx += 256) {
            int t = idx >> 7, c = idx & 127;
            float v = att[((size_t)b * Tk + t0 + t) * Ck + c];
            s_attT[c * TFRAG_S + t] = tobf(v);
        }
        __syncthreads();

        // z tile [32, 128]: wave -> (ts = wave&1, h-tiles (wave>>1), (wave>>1)+4)
        {
            const int ts = wave & 1;
            v8f az[2];
#pragma unroll
            for (int j = 0; j < 2; ++j)
                az[j] = splat8(bb[h0b + ((wave >> 1) + j * 4) * 16 + m]);
#pragma unroll 1
            for (int g = 0; g < 8; ++g) {
                v16bf a = *(const v16bf*)(s_seqh + (ts * 16 + m) * SEQH_S +
                                          g * 32 + hsel * 16);
#pragma unroll
                for (int j = 0; j < 2; ++j) {
                    int h0 = h0b + ((wave >> 1) + j * 4) * 16;
                    v16bf bf = load_b_frag_f32(
                        Wb + (size_t)(g * 32 + hsel * 16) * Hk + h0 + m, Hk);
                    az[j] = wmma_bf16(a, bf, az[j]);
                }
            }
            // store z bf16 in rep-A layout: [h_local][t fragment]
#pragma unroll
            for (int j = 0; j < 2; ++j)
#pragma unroll
                for (int r = 0; r < 8; ++r) {
                    int t    = ts * 16 + r + hsel * 8;
                    int hloc = ((wave >> 1) + j * 4) * 16 + m;
                    s_zh[hloc * TFRAG_S + a_perm(t)] = tobf(az[j][r]);
                }
        }
        __syncthreads();

        // rep += z^T-tile @ att-tile (K = 32, single WMMA k-step)
        {
            v16bf a = *(const v16bf*)(s_zh + (wave * 16 + m) * TFRAG_S + hsel * 16);
#pragma unroll
            for (int ci = 0; ci < 8; ++ci) {
                v16bf bf = *(const v16bf*)(s_attT + (ci * 16 + m) * TFRAG_S + hsel * 16);
                acc[ci] = wmma_bf16(a, bf, acc[ci]);
            }
        }
    }

    // flush rep partial (NSPLIT-way reduction via atomics)
    float* rb = rep_out + (size_t)b * Hk * Ck;
#pragma unroll
    for (int ci = 0; ci < 8; ++ci)
#pragma unroll
        for (int r = 0; r < 8; ++r) {
            int h = h0b + wave * 16 + r + hsel * 8;
            atomicAdd(&rb[(size_t)h * Ck + ci * 16 + m], acc[ci][r]);
        }
}

__global__ void zero_kernel(float* __restrict__ p, int n) {
    int i = blockIdx.x * 256 + threadIdx.x;
    if (i < n) p[i] = 0.0f;
}

__global__ __launch_bounds__(128) void head_kernel(
    const float* __restrict__ rep, const float* __restrict__ am,
    const float* __restrict__ abias, const float* __restrict__ Wt,
    const float* __restrict__ bt, float* __restrict__ logit_out,
    float* __restrict__ thres_out) {
    int b = blockIdx.x;
    int c = threadIdx.x;
    const float* rb = rep + (size_t)b * Hk * Ck + c;
    float accL = 0.0f, accT = 0.0f;
#pragma unroll 4
    for (int h = 0; h < Hk; ++h) {
        float r = rb[(size_t)h * Ck];
        accL += r * am[h * Ck + c];
        accT += r * Wt[h];
    }
    logit_out[b * Ck + c] = accL + abias[c];
    thres_out[b * Ck + c] = accT + bt[0];
}

extern "C" void kernel_launch(void* const* d_in, const int* in_sizes, int n_in,
                              void* d_out, int out_size, void* d_ws, size_t ws_size,
                              hipStream_t stream) {
    (void)in_sizes; (void)n_in; (void)out_size; (void)d_ws; (void)ws_size;
    const float* seq   = (const float*)d_in[0];
    const float* Wb    = (const float*)d_in[1];
    const float* bb    = (const float*)d_in[2];
    const float* Wa    = (const float*)d_in[3];
    const float* ba    = (const float*)d_in[4];
    const float* am    = (const float*)d_in[5];
    const float* abias = (const float*)d_in[6];
    const float* Wt    = (const float*)d_in[7];
    const float* bt    = (const float*)d_in[8];

    float* out       = (float*)d_out;
    float* att_out   = out;                                // [B,T,C]
    float* rep_out   = att_out + (size_t)Bk * Tk * Ck;     // [B,H,C]
    float* logit_out = rep_out + (size_t)Bk * Hk * Ck;     // [B,C]
    float* thres_out = logit_out + (size_t)Bk * Ck;        // [B,C]

    int repN = Bk * Hk * Ck;
    zero_kernel<<<(repN + 255) / 256, 256, 0, stream>>>(rep_out, repN);

    size_t lds1 = 40960 + 17408 + 66560;                   // ~122 KB
    stage1_attention<<<dim3(Bk * (Tk / TB)), 256, lds1, stream>>>(
        seq, Wb, bb, Wa, ba, att_out);

    size_t lds2 = 40960 + 17408 + 12288 + 12288;           // ~81 KB
    stage2_rep<<<dim3(Bk * 8 * NSPLIT), 256, lds2, stream>>>(
        seq, Wb, bb, att_out, rep_out);

    head_kernel<<<dim3(Bk), Ck, 0, stream>>>(rep_out, am, abias, Wt, bt,
                                             logit_out, thres_out);
}
